// CNN_68152541053664
// MI455X (gfx1250) — compile-verified, matrix operand order checked by reading, not run on hardware
//
#include <hip/hip_runtime.h>
#include <hip/hip_bf16.h>

// ---------------------------------------------------------------------------
// Types
// ---------------------------------------------------------------------------
typedef __bf16 bf16;
typedef __attribute__((ext_vector_type(16))) __bf16 bf16x16;
typedef __attribute__((ext_vector_type(8)))  __bf16 bf16x8;
typedef __attribute__((ext_vector_type(8)))  float  f32x8;
typedef __attribute__((ext_vector_type(4)))  unsigned int u32x4;
typedef __attribute__((ext_vector_type(8)))  int i32x8;
typedef __attribute__((ext_vector_type(4)))  int i32x4;

union FragU { bf16x16 v; bf16x8 h[2]; };

#if __has_builtin(__builtin_amdgcn_tensor_load_to_lds)
#define USE_TDM 1
#else
#define USE_TDM 0
#endif

// Staged K-chunk width (elements). All layer K dims (2400,384,768,192) divide by 96.
#define KC 96

// ---------------------------------------------------------------------------
// Fragment loads (per ISA "16-bit A-Matrix 16x32" layout: lane&15 = M row /
// N col, hi = lane>>4 selects K in [hi*8,hi*8+8) and [16+hi*8,16+hi*8+8)).
// ---------------------------------------------------------------------------
__device__ __forceinline__ bf16x16 load_frag_bf16(const bf16* __restrict__ row, int hi) {
    FragU u;
    u.h[0] = *(const bf16x8*)(row + hi * 8);
    u.h[1] = *(const bf16x8*)(row + 16 + hi * 8);
    return u.v;
}

__device__ __forceinline__ bf16x16 load_frag_f32(const float* __restrict__ row, int hi) {
    bf16x16 v;
    const float4 f0 = *(const float4*)(row + hi * 8);
    const float4 f1 = *(const float4*)(row + hi * 8 + 4);
    const float4 f2 = *(const float4*)(row + 16 + hi * 8);
    const float4 f3 = *(const float4*)(row + 16 + hi * 8 + 4);
    v[0]  = (bf16)f0.x; v[1]  = (bf16)f0.y; v[2]  = (bf16)f0.z; v[3]  = (bf16)f0.w;
    v[4]  = (bf16)f1.x; v[5]  = (bf16)f1.y; v[6]  = (bf16)f1.z; v[7]  = (bf16)f1.w;
    v[8]  = (bf16)f2.x; v[9]  = (bf16)f2.y; v[10] = (bf16)f2.z; v[11] = (bf16)f2.w;
    v[12] = (bf16)f3.x; v[13] = (bf16)f3.y; v[14] = (bf16)f3.z; v[15] = (bf16)f3.w;
    return v;
}

// ---------------------------------------------------------------------------
// TDM issue: DMA one [tileRows x tileCols] bf16 tile of W (row stride = Kelem
// elements) into LDS at byte offset lds_off. Descriptor per CDNA5 ISA §8:
//   group0: count=1 | lds_addr | global_addr[56:0] | type=2
//   group1: data_size=1(2B) | tensor_dim0=Kelem | tensor_dim1=Nrows |
//           tile_dim0=tileCols | tile_dim1=tileRows | tensor_dim0_stride=Kelem
//   groups 2/3: zero (2-D tile)
// ---------------------------------------------------------------------------
__device__ __forceinline__ void tdm_issue(const void* gsrc, unsigned lds_off,
                                          int Kelem, int Nrows, int tileRows,
                                          int tileCols) {
#if USE_TDM
    const unsigned long long ga = (unsigned long long)(size_t)gsrc;
    u32x4 g0;
    g0[0] = 1u;                                            // count = 1 (valid)
    g0[1] = lds_off;                                       // LDS byte address
    g0[2] = (unsigned)ga;                                  // global_addr[31:0]
    g0[3] = (unsigned)((ga >> 32) & 0x1ffffffu) | (2u << 30); // [56:32] | type=2
    i32x8 g1;
    g1[0] = 1 << 16;                                       // data_size = 2 bytes
    g1[1] = (Kelem & 0xffff) << 16;                        // tensor_dim0[15:0]
    g1[2] = ((Kelem >> 16) & 0xffff) | ((Nrows & 0xffff) << 16); // dim0 hi | dim1 lo
    g1[3] = ((Nrows >> 16) & 0xffff) | ((tileCols & 0xffff) << 16); // dim1 hi | tile_dim0
    g1[4] = tileRows & 0xffff;                             // tile_dim1 (tile_dim2=0)
    g1[5] = Kelem;                                         // tensor_dim0_stride lo
    g1[6] = 0;                                             // stride hi | dim1_stride
    g1[7] = 0;
    i32x4 z4 = {0, 0, 0, 0};
#if defined(__clang_major__) && (__clang_major__ >= 23)
    i32x8 z8 = {0, 0, 0, 0, 0, 0, 0, 0};
    __builtin_amdgcn_tensor_load_to_lds(g0, g1, z4, z4, z8, 0);
#else
    __builtin_amdgcn_tensor_load_to_lds(g0, g1, z4, z4, 0);
#endif
#endif
}

// Fallback staging when the TDM builtin is unavailable: cooperative
// global_load_b128 + ds_store_b128 by the whole workgroup.
__device__ __forceinline__ void stage_fallback(bf16* dst, const bf16* __restrict__ src,
                                               int Kelem, int tileRows, int tid) {
    const int cpr = KC / 8;                    // 16B chunks per row
    for (int c = tid; c < tileRows * cpr; c += 256) {
        const int row = c / cpr, part = c % cpr;
        bf16x8 v = *(const bf16x8*)(src + (size_t)row * Kelem + part * 8);
        *(bf16x8*)(dst + row * KC + part * 8) = v;
    }
}

// ---------------------------------------------------------------------------
// Weight prep: dense "effective" conv weight  W_eff[N=CO*3, K=CI*3] (bf16)
//   W_eff[co*3+lo][ci*3+li] = w[co][ci][li-lo+1]  if 0 <= li-lo+1 < 3 else 0
// ---------------------------------------------------------------------------
__global__ void build_conv_weff(const float* __restrict__ w, bf16* __restrict__ weff,
                                int CO, int CI) {
    const long long total = (long long)CO * 3 * CI * 3;
    long long idx = (long long)blockIdx.x * blockDim.x + threadIdx.x;
    if (idx >= total) return;
    const int Kd = CI * 3;
    const int n = (int)(idx / Kd);
    const int k = (int)(idx % Kd);
    const int co = n / 3, lo = n % 3;
    const int ci = k / 3, li = k % 3;
    const int kk = li - lo + 1;
    float val = 0.0f;
    if (kk >= 0 && kk < 3) val = w[((long long)co * CI + ci) * 3 + kk];
    weff[idx] = (bf16)val;
}

__global__ void cvt_f32_to_bf16(const float* __restrict__ src, bf16* __restrict__ dst,
                                long long n) {
    long long idx = (long long)blockIdx.x * blockDim.x + threadIdx.x;
    if (idx < n) dst[idx] = (bf16)src[idx];
}

// ---------------------------------------------------------------------------
// Mask precompute: per item, userAny (40-bit) and chanFull (20-bit) bitmaps.
// ---------------------------------------------------------------------------
__global__ void mask_kernel(const float* __restrict__ state,
                            unsigned long long* __restrict__ userAny,
                            unsigned int* __restrict__ chanFull, int B) {
    int b = blockIdx.x * blockDim.x + threadIdx.x;
    if (b >= B) return;
    const float* s = state + (long long)b * 2400 + 2;
    unsigned long long ua = 0ull;
    unsigned int cf = 0u;
    for (int chan = 0; chan < 20; ++chan) {
        int cnt = 0;
        #pragma unroll
        for (int user = 0; user < 40; ++user) {
            const int ss = chan * 40 + user;
            const int v = (s[ss * 3] != 0.0f) ? 1 : 0;
            cnt += v;
            ua |= (unsigned long long)v << user;
        }
        cf |= (unsigned int)(cnt >= 2) << chan;
    }
    userAny[b] = ua;
    chanFull[b] = cf;
}

// ---------------------------------------------------------------------------
// WMMA GEMM:  Out[M,N] = act( A[M,K] @ W[N,K]^T + bias )
//   - 256 threads = 8 waves; wave w owns rows [blockIdx.y*128 + w*16, +16)
//   - weight tile [NT*16 x 96] staged into LDS by TDM (double-buffered,
//     wave 0 issues, s_wait_tensorcnt + workgroup barrier publishes);
//     3 wmma K-substeps consumed per stage -> 1 barrier per 12 wmmas (NT=4)
//   - B fragments are loaded in forward order and consumed in REVERSE order:
//     the first wmma depends on the last ds_load, so all fragments stay live
//     in distinct registers and only one s_wait_dscnt is paid per burst
//   - AF32: layer-1 converts the f32 state in-flight
//   - FINAL: fuse bias + action-mask and emit f32 with -inf
// ---------------------------------------------------------------------------
template<int K, int NT, bool AF32, bool BIAS3, bool RELU, bool FINAL>
__global__ __launch_bounds__(256)
void gemm_wmma(const void* __restrict__ Aptr, const bf16* __restrict__ W,
               const float* __restrict__ bias, void* __restrict__ Out, int N,
               const float* __restrict__ state,
               const unsigned long long* __restrict__ userAny,
               const unsigned int* __restrict__ chanFull) {
    __shared__ __align__(16) bf16 smem[2][64 * KC];   // 2 x 12 KB double buffer

    const int tid  = threadIdx.x;
    const int wave = tid >> 5;
    const int lane = tid & 31;
    const int r    = lane & 15;
    const int hi   = lane >> 4;
    const int tileRows = NT * 16;

    const int mBase = blockIdx.y * 128 + wave * 16;
    const int nBase = blockIdx.x * (16 * NT);
    const int m     = mBase + r;

    f32x8 acc[NT];
    #pragma unroll
    for (int t = 0; t < NT; ++t)
        #pragma unroll
        for (int e = 0; e < 8; ++e) acc[t][e] = 0.0f;

    const float* Af = (const float*)Aptr;
    const bf16*  Ab = (const bf16*)Aptr;
    const bf16*  Wblk = W + (size_t)nBase * K;

    // Prologue: stage the first weight tile into buffer 0.
#if USE_TDM
    if (wave == 0) tdm_issue(Wblk, 0u, K, N, tileRows, KC);
#else
    stage_fallback(&smem[0][0], Wblk, K, tileRows, tid);
#endif

    int cur = 0;
    for (int ks = 0; ks < K; ks += KC) {
#if USE_TDM
        if (wave == 0) __builtin_amdgcn_s_wait_tensorcnt(0);
#endif
        __syncthreads();   // publish buf[cur]; all waves done with buf[cur^1]

        if (ks + KC < K) {
            const unsigned nextOff = (unsigned)((cur ^ 1) * sizeof(smem[0]));
#if USE_TDM
            if (wave == 0) tdm_issue(Wblk + (ks + KC), nextOff, K, N, tileRows, KC);
#else
            stage_fallback(&smem[cur ^ 1][0], Wblk + (ks + KC), K, tileRows, tid);
            (void)nextOff;
#endif
            // Warm L0 for next stage's A row (lowers to global_prefetch_b8).
            if (AF32) __builtin_prefetch(Af + (long long)m * K + ks + KC, 0, 3);
            else      __builtin_prefetch(Ab + (long long)m * K + ks + KC, 0, 3);
        }

        #pragma unroll
        for (int kk = 0; kk < KC; kk += 32) {
            bf16x16 a;
            if (AF32) a = load_frag_f32(Af + (long long)m * K + ks + kk, hi);
            else      a = load_frag_bf16(Ab + (long long)m * K + ks + kk, hi);

            // Phase 1: gather all NT B fragments from LDS (forward order).
            FragU u[NT];
            #pragma unroll
            for (int t = 0; t < NT; ++t) {
                const bf16* wp = &smem[cur][(t * 16 + r) * KC + kk];
                u[t].h[0] = *(const bf16x8*)(wp + hi * 8);
                u[t].h[1] = *(const bf16x8*)(wp + 16 + hi * 8);
            }
            // Phase 2: wmma burst in REVERSE order -> first wmma waits on the
            // last ds_load; remaining wmmas consume already-resident fragments.
            #pragma unroll
            for (int t = NT - 1; t >= 0; --t) {
                acc[t] = __builtin_amdgcn_wmma_f32_16x16x32_bf16(
                    /*neg_a=*/false, a, /*neg_b=*/false, u[t].v,
                    /*c_mod=*/(short)0, acc[t], /*reuse_a=*/false, /*reuse_b=*/false);
            }
        }
        cur ^= 1;
    }

    // Epilogue. C/D layout: VGPR v -> row mBase + v + 8*hi, col nBase + t*16 + r.
    #pragma unroll
    for (int t = 0; t < NT; ++t) {
        const int col = nBase + t * 16 + r;
        const float bv = BIAS3 ? bias[col / 3] : bias[col];
        #pragma unroll
        for (int v = 0; v < 8; ++v) {
            const int row = mBase + v + 8 * hi;
            float val = acc[t][v] + bv;
            if (RELU) val = fmaxf(val, 0.0f);
            if (FINAL) {
                const int user = col % 40;
                const int chan = col / 40;
                const bool vis = state[(long long)row * 2400 + col * 3 + 2] != 0.0f;
                const bool masked = vis
                    || ((userAny[row] >> user) & 1ull)
                    || ((chanFull[row] >> chan) & 1u);
                ((float*)Out)[(long long)row * N + col] =
                    masked ? -__builtin_huge_valf() : val;
            } else {
                ((bf16*)Out)[(long long)row * N + col] = (bf16)val;
            }
        }
    }
}

// ---------------------------------------------------------------------------
// Host side
// ---------------------------------------------------------------------------
extern "C" void kernel_launch(void* const* d_in, const int* in_sizes, int n_in,
                              void* d_out, int out_size, void* d_ws, size_t ws_size,
                              hipStream_t stream) {
    const int B = 16384;

    const float* state = (const float*)d_in[0];
    const float* w1  = (const float*)d_in[1];
    const float* b1  = (const float*)d_in[2];
    const float* w2  = (const float*)d_in[3];
    const float* b2  = (const float*)d_in[4];
    const float* w3  = (const float*)d_in[5];
    const float* b3  = (const float*)d_in[6];
    const float* wl1 = (const float*)d_in[7];
    const float* bl1 = (const float*)d_in[8];
    const float* wl2 = (const float*)d_in[9];
    const float* bl2 = (const float*)d_in[10];

    // workspace bump allocator (total ~60 MB; intermediates stay L2-resident)
    char* ws = (char*)d_ws;
    auto alloc = [&](size_t bytes) -> char* {
        char* p = ws;
        ws += (bytes + 255) & ~(size_t)255;
        return p;
    };
    bf16* W1e = (bf16*)alloc((size_t)384 * 2400 * sizeof(bf16));
    bf16* W2e = (bf16*)alloc((size_t)768 * 384  * sizeof(bf16));
    bf16* W3e = (bf16*)alloc((size_t)384 * 768  * sizeof(bf16));
    bf16* WL1 = (bf16*)alloc((size_t)192 * 384  * sizeof(bf16));
    bf16* WL2 = (bf16*)alloc((size_t)800 * 192  * sizeof(bf16));
    bf16* Y1  = (bf16*)alloc((size_t)B * 384 * sizeof(bf16));
    bf16* Y2  = (bf16*)alloc((size_t)B * 768 * sizeof(bf16));
    bf16* Y3  = (bf16*)alloc((size_t)B * 384 * sizeof(bf16));
    bf16* Y4  = (bf16*)alloc((size_t)B * 192 * sizeof(bf16));
    unsigned long long* userAny  = (unsigned long long*)alloc((size_t)B * 8);
    unsigned int*       chanFull = (unsigned int*)alloc((size_t)B * 4);

    const int T = 256;
    auto blocks = [](long long n, int t) { return (unsigned)((n + t - 1) / t); };

    // 1) effective weights (bf16)
    build_conv_weff<<<blocks((long long)384 * 2400, T), T, 0, stream>>>(w1, W1e, 128, 800);
    build_conv_weff<<<blocks((long long)768 * 384,  T), T, 0, stream>>>(w2, W2e, 256, 128);
    build_conv_weff<<<blocks((long long)384 * 768,  T), T, 0, stream>>>(w3, W3e, 128, 256);
    cvt_f32_to_bf16<<<blocks((long long)192 * 384,  T), T, 0, stream>>>(wl1, WL1, (long long)192 * 384);
    cvt_f32_to_bf16<<<blocks((long long)800 * 192,  T), T, 0, stream>>>(wl2, WL2, (long long)800 * 192);

    // 2) action-mask bitmaps
    mask_kernel<<<blocks(B, T), T, 0, stream>>>(state, userAny, chanFull, B);

    // 3) GEMM chain (M=16384; grid = (N/(16*NT), M/128))
    {   // L1: conv1  [B,2400] f32 -> [B,384]
        dim3 g(384 / (16 * 4), B / 128);
        gemm_wmma<2400, 4, true,  true,  true,  false><<<g, T, 0, stream>>>(
            state, W1e, b1, Y1, 384, nullptr, nullptr, nullptr);
    }
    {   // L2: conv2  [B,384] -> [B,768]
        dim3 g(768 / (16 * 4), B / 128);
        gemm_wmma<384,  4, false, true,  true,  false><<<g, T, 0, stream>>>(
            Y1, W2e, b2, Y2, 768, nullptr, nullptr, nullptr);
    }
    {   // L3: conv3  [B,768] -> [B,384]
        dim3 g(384 / (16 * 4), B / 128);
        gemm_wmma<768,  4, false, true,  true,  false><<<g, T, 0, stream>>>(
            Y2, W3e, b3, Y3, 384, nullptr, nullptr, nullptr);
    }
    {   // L4: linear1 [B,384] -> [B,192]
        dim3 g(192 / (16 * 4), B / 128);
        gemm_wmma<384,  4, false, false, true,  false><<<g, T, 0, stream>>>(
            Y3, WL1, bl1, Y4, 192, nullptr, nullptr, nullptr);
    }
    {   // L5: linear2 + mask -> f32 out [B,800]
        dim3 g(800 / (16 * 2), B / 128);
        gemm_wmma<192,  2, false, false, false, true><<<g, T, 0, stream>>>(
            Y4, WL2, bl2, d_out, 800, state, userAny, chanFull);
    }
}
